// RelativeDecoderLayer_17927193494193
// MI455X (gfx1250) — matrix operand order repaired and checked by reading, not compiled, not run on hardware
//
#include <hip/hip_runtime.h>
#include <hip/hip_bf16.h>
#include <math.h>

// Problem constants (match reference)
#define Bn  8
#define Sn  1024
#define Dn  512
#define Hn  8
#define DHn 64
#define DFn 2048

typedef __attribute__((ext_vector_type(16))) _Float16 v16h;
typedef __attribute__((ext_vector_type(2)))  __fp16   v2hp;   // cvt_pkrtz return type
typedef __attribute__((ext_vector_type(8)))  float    v8f;

struct Frag { union { unsigned u[8]; v16h v; }; };

static __device__ __forceinline__ unsigned pack2f16(float a, float b) {
    union { v2hp h; unsigned u; } t;
    t.h = __builtin_amdgcn_cvt_pkrtz(a, b);   // v_cvt_pk_rtz_f16_f32
    return t.u;
}

static __device__ __forceinline__ v8f wmma_f16(const Frag& a, const Frag& b, v8f c) {
    // D = A(16x32 f16) * B(32x16 f16) + C(f32), wave32
    return __builtin_amdgcn_wmma_f32_16x16x32_f16(
        false, a.v, false, b.v, (short)0, c, false, false);
}

// A-fragment K index for packed pair j (0..7): lanes0-15 K=0..7,16..23 ; lanes16-31 K=8..15,24..31
static __device__ __forceinline__ int kkA(int j, int half) {
    return (j >> 2) * 16 + half * 8 + ((j & 3) << 1);
}
// B-fragment K index: lanes0-15 K=0..7,16..23 ; lanes16-31 K=8..15,24..31
static __device__ __forceinline__ int kkB(int j, int half) {
    return (j >> 2) * 16 + half * 8 + ((j & 3) << 1);
}

// ---------------------------------------------------------------------------
// Batched GEMM: C = act(A * B + bias), f32 in/out, f16 WMMA compute.
// REQUIRES: M % 128 == 0, N % 64 == 0, K % 32 == 0 (true for every GEMM here).
// z-batch: Aoff = z*sA ; Boff = (z/hDiv)*sB0 + (z%hDiv)*sB1 ; Coff likewise.
// B is row-major [K][N].
// ---------------------------------------------------------------------------
#define BM 128
#define BN 64
#define BK 32
#define LDSP 36   // f16 row pitch: 72B -> 4/8B-aligned packed accesses, bank-spread

__global__ __launch_bounds__(256)
void wmma_gemm_kernel(const float* __restrict__ A, int lda, long long sA,
                      const float* __restrict__ Bp, int ldb, long long sB0, long long sB1,
                      float* __restrict__ C, int ldc, long long sC0, long long sC1,
                      int K, int hDiv,
                      const float* __restrict__ bias, int relu)
{
    __shared__ _Float16 As[BM][LDSP];
    __shared__ _Float16 Bs[BN][LDSP];   // transposed: Bs[n][k]

    const int z  = blockIdx.z;
    const int zb = z / hDiv, zh = z % hDiv;
    const float* Ab = A  + (long long)z  * sA;
    const float* Bb = Bp + (long long)zb * sB0 + (long long)zh * sB1;
    float*       Cb = C  + (long long)zb * sC0 + (long long)zh * sC1;

    const int m0 = blockIdx.y * BM;
    const int n0 = blockIdx.x * BN;

    const int t    = threadIdx.x;
    const int lane = t & 31, half = lane >> 4, lrow = lane & 15;
    const int wid  = t >> 5;
    const int wm   = (wid >> 1) * 32;   // 4 wave-rows of 32
    const int wn   = (wid & 1)  * 32;   // 2 wave-cols of 32

    // A staging: thread -> row (t>>1) in [0,128), 16 consecutive k at (t&1)*16
    const int arow = t >> 1, acol = (t & 1) * 16;
    // B staging: thread -> k row (t>>3) in [0,32), 8 consecutive n at (t&7)*8
    const int bk = t >> 3, bn0 = (t & 7) * 8;

    const float* aPtr = Ab + (size_t)(m0 + arow) * lda + acol;
    const float* bPtr = Bb + (size_t)bk * ldb + n0 + bn0;

    // Software pipeline: stage tile k0=0 into registers
    float4 ar[4], br[2];
    #pragma unroll
    for (int e = 0; e < 4; ++e) ar[e] = *(const float4*)(aPtr + 4 * e);
    #pragma unroll
    for (int e = 0; e < 2; ++e) br[e] = *(const float4*)(bPtr + 4 * e);

    v8f acc[2][2] = {};

    for (int k0 = 0; k0 < K; k0 += BK) {
        __syncthreads();   // previous iteration's LDS reads complete

        // Regs -> LDS (f32 -> f16 packed)
        #pragma unroll
        for (int e = 0; e < 4; ++e) {
            union { v2hp h2[2]; uint2 u; } w;
            w.h2[0] = __builtin_amdgcn_cvt_pkrtz(ar[e].x, ar[e].y);
            w.h2[1] = __builtin_amdgcn_cvt_pkrtz(ar[e].z, ar[e].w);
            *(uint2*)&As[arow][acol + 4 * e] = w.u;
        }
        #pragma unroll
        for (int e = 0; e < 2; ++e) {
            Bs[bn0 + 4 * e + 0][bk] = (_Float16)br[e].x;
            Bs[bn0 + 4 * e + 1][bk] = (_Float16)br[e].y;
            Bs[bn0 + 4 * e + 2][bk] = (_Float16)br[e].z;
            Bs[bn0 + 4 * e + 3][bk] = (_Float16)br[e].w;
        }
        __syncthreads();

        // Issue next tile's global loads now; wait lands at next LDS store,
        // overlapping HBM latency with the WMMAs below.
        const int kn = k0 + BK;
        if (kn < K) {
            #pragma unroll
            for (int e = 0; e < 4; ++e) ar[e] = *(const float4*)(aPtr + kn + 4 * e);
            #pragma unroll
            for (int e = 0; e < 2; ++e) br[e] = *(const float4*)(bPtr + (size_t)kn * ldb + 4 * e);
        }

        // Fragments + 4 WMMAs
        Frag af[2], bf[2];
        #pragma unroll
        for (int tm = 0; tm < 2; ++tm) {
            int row = wm + tm * 16 + lrow;
            #pragma unroll
            for (int j = 0; j < 8; ++j)
                af[tm].u[j] = *(const unsigned*)&As[row][kkA(j, half)];
        }
        #pragma unroll
        for (int tn = 0; tn < 2; ++tn) {
            int col = wn + tn * 16 + lrow;
            #pragma unroll
            for (int j = 0; j < 8; ++j)
                bf[tn].u[j] = *(const unsigned*)&Bs[col][kkB(j, half)];
        }
        #pragma unroll
        for (int tm = 0; tm < 2; ++tm)
            #pragma unroll
            for (int tn = 0; tn < 2; ++tn)
                acc[tm][tn] = wmma_f16(af[tm], bf[tn], acc[tm][tn]);
    }

    // Epilogue (no guards: dims divide tiles exactly)
    #pragma unroll
    for (int tm = 0; tm < 2; ++tm)
        #pragma unroll
        for (int tn = 0; tn < 2; ++tn) {
            const int gn = n0 + wn + tn * 16 + lrow;
            const float bn = bias ? bias[gn] : 0.0f;
            #pragma unroll
            for (int vi = 0; vi < 8; ++vi) {
                int gm = m0 + wm + tm * 16 + vi + half * 8;
                float vv = acc[tm][tn][vi] + bn;
                if (relu) vv = fmaxf(vv, 0.0f);
                Cb[(size_t)gm * ldc + gn] = vv;
            }
        }
}

// ---------------------------------------------------------------------------
// Fused relative-attention scores (pre-softmax), written into the attn output.
// scores[bh,i,j] = 0.125 * ( (q_i+u)·k_j + (q_i+v)·r_{j-i+S-1} ), masked j>i.
// One wave per 16x16 (it,jt) tile; only jt<=it computed (rest fully masked).
// ---------------------------------------------------------------------------
__global__ __launch_bounds__(32)
void rel_scores_kernel(const float* __restrict__ q, const float* __restrict__ kmat,
                       const float* __restrict__ r, const float* __restrict__ u_bias,
                       const float* __restrict__ v_bias, float* __restrict__ scores)
{
    const int jt = blockIdx.x, it = blockIdx.y, bh = blockIdx.z;
    if (jt > it) return;   // fully masked tile: softmax writes zeros there
    const int b = bh / Hn, h = bh % Hn;

    const float* qb = q    + (size_t)b * Sn * Dn + h * DHn;
    const float* kb = kmat + (size_t)b * Sn * Dn + h * DHn;
    const float* rb = r    + h * DHn;
    const float* ub = u_bias + h * DHn;
    const float* vb = v_bias + h * DHn;

    const int lane = threadIdx.x & 31, half = lane >> 4, lrow = lane & 15;

    // A fragments (q+u), (q+v): rows i = it*16 + lrow ; K = DH = 64 -> 2 k-steps
    Frag aU[2], aV[2], bK[2];
    const int iq = it * 16 + lrow;
    const int jj = jt * 16 + lrow;
    #pragma unroll
    for (int ks = 0; ks < 2; ++ks) {
        #pragma unroll
        for (int j = 0; j < 8; ++j) {
            int ka = ks * 32 + kkA(j, half);
            float2 qv = *(const float2*)&qb[(size_t)iq * Dn + ka];
            aU[ks].u[j] = pack2f16(qv.x + ub[ka], qv.y + ub[ka + 1]);
            aV[ks].u[j] = pack2f16(qv.x + vb[ka], qv.y + vb[ka + 1]);
            int kbb = ks * 32 + kkB(j, half);
            float2 kv = *(const float2*)&kb[(size_t)jj * Dn + kbb];
            bK[ks].u[j] = pack2f16(kv.x, kv.y);
        }
    }

    // content term
    v8f acc_ac = {};
    acc_ac = wmma_f16(aU[0], bK[0], acc_ac);
    acc_ac = wmma_f16(aU[1], bK[1], acc_ac);

    // position term over the p-band [pbase, pbase+32), p = j-i+S-1; clamp the
    // address and zero the data (branchless -> EXEC stays all-ones).
    __shared__ float bdS[16][33];
    const int pbase = (jt - it) * 16 + Sn - 1 - 16;
    #pragma unroll
    for (int ct = 0; ct < 2; ++ct) {
        const int p  = pbase + ct * 16 + lrow;
        const int pc = min(max(p, 0), Sn - 1);
        const float msk = (p == pc) ? 1.0f : 0.0f;
        Frag bR[2];
        #pragma unroll
        for (int ks = 0; ks < 2; ++ks)
            #pragma unroll
            for (int j = 0; j < 8; ++j) {
                int kbb = ks * 32 + kkB(j, half);
                float2 rv = *(const float2*)&rb[(size_t)pc * Dn + kbb];
                bR[ks].u[j] = pack2f16(rv.x * msk, rv.y * msk);
            }
        v8f acc_bd = {};
        acc_bd = wmma_f16(aV[0], bR[0], acc_bd);
        acc_bd = wmma_f16(aV[1], bR[1], acc_bd);
        #pragma unroll
        for (int vi = 0; vi < 8; ++vi)
            bdS[vi + half * 8][ct * 16 + lrow] = acc_bd[vi];
    }
    __syncthreads();

    // combine with rel-shift: bd column c = (n - m) + 16
    #pragma unroll
    for (int vi = 0; vi < 8; ++vi) {
        int m  = vi + half * 8;
        int ig = it * 16 + m;
        int jg = jt * 16 + lrow;
        float s = (acc_ac[vi] + bdS[m][lrow - m + 16]) * 0.125f;   // 1/sqrt(64)
        if (jg > ig) s = -1e9f;
        scores[((size_t)bh * Sn + ig) * Sn + jg] = s;
    }
}

// ---------------------------------------------------------------------------
// In-place causal row softmax: single pass, row resident in registers.
// attn[row, j] = softmax over j<=i ; exact 0 for j>i (never reads j>i memory).
// ---------------------------------------------------------------------------
__global__ __launch_bounds__(256)
void softmax_kernel(float* __restrict__ attn)
{
    __shared__ float red[256];
    const int row = blockIdx.x;          // bh*S + i
    const int i   = row % Sn;
    float* base = attn + (size_t)row * Sn;
    const int tid = threadIdx.x;

    float vals[4];
    float m = -3.0e38f;
    #pragma unroll
    for (int c = 0; c < 4; ++c) {
        const int j = tid + c * 256;
        float v = (j <= i) ? base[j] : -3.0e38f;   // never trust j>i memory
        vals[c] = v;
        m = fmaxf(m, v);
    }
    red[tid] = m; __syncthreads();
    for (int s = 128; s > 0; s >>= 1) { if (tid < s) red[tid] = fmaxf(red[tid], red[tid + s]); __syncthreads(); }
    const float mx = red[0]; __syncthreads();

    float sum = 0.0f;
    #pragma unroll
    for (int c = 0; c < 4; ++c) {
        const int j = tid + c * 256;
        float e = (j <= i) ? __expf(vals[c] - mx) : 0.0f;
        vals[c] = e;
        sum += e;
    }
    red[tid] = sum; __syncthreads();
    for (int s = 128; s > 0; s >>= 1) { if (tid < s) red[tid] += red[tid + s]; __syncthreads(); }
    const float inv = 1.0f / red[0];

    #pragma unroll
    for (int c = 0; c < 4; ++c)
        base[tid + c * 256] = vals[c] * inv;
}

// ---------------------------------------------------------------------------
// out = LN(a + b) * g + be   (D = 512, block = 256 -> 2 elems/thread)
// ---------------------------------------------------------------------------
__global__ __launch_bounds__(256)
void ln_kernel(const float* __restrict__ a, const float* __restrict__ b,
               const float* __restrict__ g, const float* __restrict__ be,
               float* __restrict__ outp)
{
    __shared__ float red[256];
    const int row = blockIdx.x, tid = threadIdx.x;
    const size_t base = (size_t)row * Dn;

    float v0 = a[base + tid]       + b[base + tid];
    float v1 = a[base + tid + 256] + b[base + tid + 256];
    red[tid] = v0 + v1; __syncthreads();
    for (int s = 128; s > 0; s >>= 1) { if (tid < s) red[tid] += red[tid + s]; __syncthreads(); }
    const float mu = red[0] / (float)Dn; __syncthreads();

    float d0 = v0 - mu, d1 = v1 - mu;
    red[tid] = d0 * d0 + d1 * d1; __syncthreads();
    for (int s = 128; s > 0; s >>= 1) { if (tid < s) red[tid] += red[tid + s]; __syncthreads(); }
    const float rs = rsqrtf(red[0] / (float)Dn + 1e-5f);

    outp[base + tid]       = d0 * rs * g[tid]       + be[tid];
    outp[base + tid + 256] = d1 * rs * g[tid + 256] + be[tid + 256];
}

// ---------------------------------------------------------------------------
extern "C" void kernel_launch(void* const* d_in, const int* in_sizes, int n_in,
                              void* d_out, int out_size, void* d_ws, size_t ws_size,
                              hipStream_t stream)
{
    (void)in_sizes; (void)n_in; (void)out_size; (void)ws_size;

    const float* x    = (const float*)d_in[0];
    const float* rel  = (const float*)d_in[1];
    /* d_in[2] = mask (causal triu k=1) is implicit in the kernels */
    const float* Wq   = (const float*)d_in[3];
    const float* Wk   = (const float*)d_in[4];
    const float* Wv   = (const float*)d_in[5];
    const float* Wr   = (const float*)d_in[6];
    const float* Wo   = (const float*)d_in[7];
    const float* u_b  = (const float*)d_in[8];
    const float* v_b  = (const float*)d_in[9];
    const float* ln1g = (const float*)d_in[10];
    const float* ln1b = (const float*)d_in[11];
    const float* W1   = (const float*)d_in[12];
    const float* b1   = (const float*)d_in[13];
    const float* W2   = (const float*)d_in[14];
    const float* b2   = (const float*)d_in[15];
    const float* ln2g = (const float*)d_in[16];
    const float* ln2b = (const float*)d_in[17];

    float* out  = (float*)d_out;
    float* attn = out + (size_t)Bn * Sn * Dn;   // output tuple: (out, attn)

    // Workspace layout (~103 MB of f32); ff1/ff2 overlay dead buffers.
    float* ws = (float*)d_ws;
    const size_t NQ = (size_t)Bn * Sn * Dn;  // 4,194,304
    const size_t SD = (size_t)Sn * Dn;       //   524,288
    float* q   = ws;
    float* kbf = ws + NQ;
    float* vbf = ws + 2 * NQ;
    float* rbf = ws + 3 * NQ;
    float* ctx = ws + 3 * NQ + SD;
    float* o   = ctx + NQ;
    float* h1  = o + NQ;
    float* ff1 = ws;   // overlays q/k/v/r/ctx (dead by then), 16.8M floats
    float* ff2 = o;    // overlays o (dead after ln1)

    const dim3 blk(256);
    const int  MB = Bn * Sn;   // 8192
    auto grid2 = [](int M, int N, int Z) {
        return dim3((unsigned)(N / BN), (unsigned)(M / BM), (unsigned)Z);
    };

    // QKV + R projections
    wmma_gemm_kernel<<<grid2(MB, Dn, 1), blk, 0, stream>>>(
        x, Dn, 0, Wq, Dn, 0, 0, q, Dn, 0, 0, Dn, 1, nullptr, 0);
    wmma_gemm_kernel<<<grid2(MB, Dn, 1), blk, 0, stream>>>(
        x, Dn, 0, Wk, Dn, 0, 0, kbf, Dn, 0, 0, Dn, 1, nullptr, 0);
    wmma_gemm_kernel<<<grid2(MB, Dn, 1), blk, 0, stream>>>(
        x, Dn, 0, Wv, Dn, 0, 0, vbf, Dn, 0, 0, Dn, 1, nullptr, 0);
    wmma_gemm_kernel<<<grid2(Sn, Dn, 1), blk, 0, stream>>>(
        rel, Dn, 0, Wr, Dn, 0, 0, rbf, Dn, 0, 0, Dn, 1, nullptr, 0);

    // Scores (content + shifted position term) -> attn region of d_out
    rel_scores_kernel<<<dim3(Sn / 16, Sn / 16, Bn * Hn), dim3(32), 0, stream>>>(
        q, kbf, rbf, u_b, v_b, attn);

    // Causal softmax in place
    softmax_kernel<<<dim3(Bn * Hn * Sn), blk, 0, stream>>>(attn);

    // ctx[b,:,h,:] = attn[b,h] @ v[b,:,h,:]   (batched over b*h)
    wmma_gemm_kernel<<<grid2(Sn, DHn, Bn * Hn), blk, 0, stream>>>(
        attn, Sn, (long long)Sn * Sn,
        vbf, Dn, (long long)Sn * Dn, (long long)DHn,
        ctx, Dn, (long long)Sn * Dn, (long long)DHn,
        Sn, Hn, nullptr, 0);

    // Output projection
    wmma_gemm_kernel<<<grid2(MB, Dn, 1), blk, 0, stream>>>(
        ctx, Dn, 0, Wo, Dn, 0, 0, o, Dn, 0, 0, Dn, 1, nullptr, 0);

    // h1 = LN(x + o)
    ln_kernel<<<dim3(MB), blk, 0, stream>>>(x, o, ln1g, ln1b, h1);

    // FFN
    wmma_gemm_kernel<<<grid2(MB, DFn, 1), blk, 0, stream>>>(
        h1, Dn, 0, W1, DFn, 0, 0, ff1, DFn, 0, 0, Dn, 1, b1, 1);
    wmma_gemm_kernel<<<grid2(MB, Dn, 1), blk, 0, stream>>>(
        ff1, DFn, 0, W2, Dn, 0, 0, ff2, Dn, 0, 0, DFn, 1, b2, 0);

    // out = LN(h1 + ff2)
    ln_kernel<<<dim3(MB), blk, 0, stream>>>(h1, ff2, ln2g, ln2b, out);
}
